// MultiHeadAttention_6536940225097
// MI455X (gfx1250) — compile-verified, matrix operand order checked by reading, not run on hardware
//
#include <hip/hip_runtime.h>

typedef __attribute__((ext_vector_type(16))) _Float16 v16h;
typedef __attribute__((ext_vector_type(8)))  float    v8f;

#define NHEAD  16
#define DMODEL 1024
#define HD     64
#define BATCH  2
#define SEQ    2048
#define MTOT   (BATCH*SEQ)
#define MWORDS (SEQ/32)   // 64 packed mask words per query row

// Fragment layout for V_WMMA_F32_16X16X32_F16 (wave32):
//   A: m = lane&15, half e -> k = 16*(e>>3) + 8*(lane>>4) + (e&7)
//   B: n = lane&15, same k mapping
//   C/D: row = e + 8*(lane>>4), col = lane&15
// So a fragment is two contiguous 16-byte chunks at +8*kgrp and +16+8*kgrp halves.
__device__ __forceinline__ v16h load_frag(const _Float16* p, int kgrp) {
  union { uint4 u[2]; v16h h; } c;
  c.u[0] = *(const uint4*)(p + 8 * kgrp);
  c.u[1] = *(const uint4*)(p + 16 + 8 * kgrp);
  return c.h;
}

__device__ __forceinline__ v8f wmma_f16(v16h a, v16h b, v8f c) {
  return __builtin_amdgcn_wmma_f32_16x16x32_f16(false, a, false, b, (short)0, c,
                                                false, false);
}

// DPP16 row-rotate (within 16-lane row) — VALU-pipe cross-lane reduction,
// no LDS/bpermute traffic. CTRL: ROW_ROR:n = 0x120 + n.
template <int CTRL>
__device__ __forceinline__ float dpp_mov(float x) {
  return __builtin_bit_cast(
      float, __builtin_amdgcn_update_dpp(0, __builtin_bit_cast(int, x), CTRL,
                                         0xf, 0xf, true));
}
__device__ __forceinline__ float row16_max(float x) {
  x = fmaxf(x, dpp_mov<0x121>(x));  // ror 1
  x = fmaxf(x, dpp_mov<0x122>(x));  // ror 2
  x = fmaxf(x, dpp_mov<0x124>(x));  // ror 4
  x = fmaxf(x, dpp_mov<0x128>(x));  // ror 8
  return x;
}
__device__ __forceinline__ float row16_sum(float x) {
  x += dpp_mov<0x121>(x);
  x += dpp_mov<0x122>(x);
  x += dpp_mov<0x124>(x);
  x += dpp_mov<0x128>(x);
  return x;
}

// ---------------------------------------------------------------------------
// Pack bool mask bytes (0/1) into bit words: mw[b][q][64], bit k = mask[b][q][k].
// ---------------------------------------------------------------------------
__global__ __launch_bounds__(256) void mask_pack_kernel(
    const unsigned char* __restrict__ m, unsigned* __restrict__ mw) {
  size_t i = (size_t)blockIdx.x * 256 + threadIdx.x;  // word index
  const unsigned* p32 = (const unsigned*)(m + i * 32);
  unsigned w = 0;
#pragma unroll
  for (int j = 0; j < 8; ++j) {
    unsigned v = p32[j];  // 4 bool bytes, each 0 or 1
    w |= ((v & 1u) | ((v >> 7) & 2u) | ((v >> 14) & 4u) | ((v >> 21) & 8u))
         << (j * 4);
  }
  mw[i] = w;
}

// ---------------------------------------------------------------------------
// GEMM kernel, 128x64 block tile, BK=32, 8 waves (wave32) per block.
// MODE 0: Q/K head projection. A=f32 [4096,1024], W=f32 [16][1024][64],
//         out f16 [h][b][l][64] (blockIdx.y == head)
// MODE 1: V head projection, out stored TRANSPOSED f16 [h][b][64][l]
// MODE 2: output projection. A=f16 attn-concat [4096,1024], W=f32 [1024,1024],
//         out f32 = acc + bias + residual
// ---------------------------------------------------------------------------
template <int MODE>
__global__ __launch_bounds__(256) void proj_kernel(
    const float* __restrict__ Af, const _Float16* __restrict__ Ah,
    const float* __restrict__ W, _Float16* __restrict__ outH,
    float* __restrict__ outF, const float* __restrict__ bias,
    const float* __restrict__ resid) {
  constexpr int BM = 128, BN = 64, BK = 32, AST = 40, BST = 40;
  alignas(16) __shared__ _Float16 As[BM * AST];  // [row][k] padded
  alignas(16) __shared__ _Float16 Bs[BN * BST];  // [n][k]  padded (transposed)

  const int tid = threadIdx.x;
  const int wave = tid >> 5, lane = tid & 31, lo = lane & 15, hi = lane >> 4;
  const int rb = blockIdx.x * BM;
  const int nb = blockIdx.y * BN;

  v8f acc[4] = {};

  for (int k0 = 0; k0 < DMODEL; k0 += BK) {
    // stage A tile (convert f32->f16 for modes 0/1)
    for (int i = tid; i < BM * BK; i += 256) {
      int row = i >> 5, kk = i & 31;
      float v = (MODE == 2) ? (float)Ah[(size_t)(rb + row) * DMODEL + k0 + kk]
                            : Af[(size_t)(rb + row) * DMODEL + k0 + kk];
      As[row * AST + kk] = (_Float16)v;
    }
    // stage B tile transposed into [n][k]
    for (int i = tid; i < BK * BN; i += 256) {
      int kk = i >> 6, n = i & 63;
      float v = (MODE == 2)
                    ? W[(size_t)(k0 + kk) * DMODEL + nb + n]
                    : W[((size_t)blockIdx.y * DMODEL + (k0 + kk)) * HD + n];
      Bs[n * BST + kk] = (_Float16)v;
    }
    __syncthreads();

    v16h a = load_frag(&As[(wave * 16 + lo) * AST], hi);
#pragma unroll
    for (int t = 0; t < 4; ++t) {
      v16h b = load_frag(&Bs[(t * 16 + lo) * BST], hi);
      acc[t] = wmma_f16(a, b, acc[t]);
    }
    __syncthreads();
  }

#pragma unroll
  for (int t = 0; t < 4; ++t) {
#pragma unroll
    for (int e = 0; e < 8; ++e) {
      int r = rb + wave * 16 + e + 8 * hi;  // global M row
      int feat = t * 16 + lo;               // col within 64-wide N tile
      float val = acc[t][e];
      if (MODE == 0) {
        int bb = r >> 11, l = r & (SEQ - 1);
        outH[(((size_t)blockIdx.y * BATCH + bb) * SEQ + l) * HD + feat] =
            (_Float16)val;
      } else if (MODE == 1) {
        int bb = r >> 11, l = r & (SEQ - 1);
        outH[(((size_t)blockIdx.y * BATCH + bb) * HD + feat) * SEQ + l] =
            (_Float16)val;
      } else {
        int col = nb + feat;
        outF[(size_t)r * DMODEL + col] =
            val + bias[col] + resid[(size_t)r * DMODEL + col];
      }
    }
  }
}

// ---------------------------------------------------------------------------
// Flash attention: grid (16 qblocks, 16 heads, 2 batch), 8 waves/block.
// Each wave owns 16 query rows; streams keys 32 at a time, mask words
// prefetched 128 keys at a time (one uint4 per row).
// scores = (Q Kh^T) * 0.125, mask bit -> -inf, online softmax, O += P*V.
// Row reductions use DPP16 row_ror (VALU pipe, no LDS traffic).
// ---------------------------------------------------------------------------
__global__ __launch_bounds__(256) void flash_kernel(
    const _Float16* __restrict__ qh, const _Float16* __restrict__ kh,
    const _Float16* __restrict__ vt, const unsigned* __restrict__ maskw,
    _Float16* __restrict__ attn) {
  alignas(16) __shared__ _Float16 Plds[8 * 16 * 40];  // per-wave 16x32 P stage

  const int tid = threadIdx.x;
  const int wave = tid >> 5, lane = tid & 31, lo = lane & 15, hi = lane >> 4;
  const int h = blockIdx.y, b = blockIdx.z;
  const int q0 = blockIdx.x * 128 + wave * 16;
  const size_t hb = (size_t)h * BATCH + b;

  // Q fragments stay resident: A-layout direct from global (two 16B loads each)
  const _Float16* Qbase = qh + (hb * SEQ + q0 + lo) * HD;
  v16h aQ0 = load_frag(Qbase, hi);       // features 0..31
  v16h aQ1 = load_frag(Qbase + 32, hi);  // features 32..63

  v8f O[4] = {};
  float mrun[8], lrun[8];
#pragma unroll
  for (int e = 0; e < 8; ++e) {
    mrun[e] = -__builtin_inff();
    lrun[e] = 0.f;
  }

  for (int kb0 = 0; kb0 < SEQ; kb0 += 128) {
    // one uint4 of mask bits per row covers keys kb0..kb0+127
    unsigned mword[8][4];
#pragma unroll
    for (int e = 0; e < 8; ++e) {
      *(uint4*)mword[e] = *(const uint4*)(
          maskw + ((size_t)b * SEQ + (q0 + e + 8 * hi)) * MWORDS + (kb0 >> 5));
    }
#pragma unroll
    for (int c = 0; c < 4; ++c) {
      const int kb = kb0 + c * 32;
      // ---- scores S (16x32) = Q(16x64) x K^T -----------------------------
      v8f S[2];
#pragma unroll
      for (int t = 0; t < 2; ++t) {
        const _Float16* Kbase = kh + (hb * SEQ + kb + t * 16 + lo) * HD;
        v8f s = {};
        s = wmma_f16(aQ1, load_frag(Kbase + 32, hi), s);
        s = wmma_f16(aQ0, load_frag(Kbase, hi), s);
        S[t] = s;
      }
      // ---- scale + mask bits + online softmax ----------------------------
#pragma unroll
      for (int e = 0; e < 8; ++e) {
        const unsigned w = mword[e][c];
        float s0 = S[0][e] * 0.125f;
        float s1 = S[1][e] * 0.125f;
        if ((w >> lo) & 1u) s0 = -__builtin_inff();
        if ((w >> (lo + 16)) & 1u) s1 = -__builtin_inff();
        float mx = row16_max(fmaxf(s0, s1));
        float mnew = fmaxf(mrun[e], mx);
        float alpha = __expf(mrun[e] - mnew);
        float p0 = __expf(s0 - mnew);
        float p1 = __expf(s1 - mnew);
        float ps = row16_sum(p0 + p1);
        lrun[e] = lrun[e] * alpha + ps;
        mrun[e] = mnew;
        // redistribute P: C-layout -> LDS rows (private per-wave slice)
        _Float16* pr = &Plds[(wave * 16 + e + 8 * hi) * 40];
        pr[lo] = (_Float16)p0;
        pr[16 + lo] = (_Float16)p1;
#pragma unroll
        for (int t = 0; t < 4; ++t) O[t][e] *= alpha;
      }
      // ---- O += P(16x32) x V(32x64); V transposed -> contiguous B --------
      v16h aP = load_frag(&Plds[(wave * 16 + lo) * 40], hi);
#pragma unroll
      for (int t = 0; t < 4; ++t) {
        const _Float16* Vbase = vt + (hb * HD + t * 16 + lo) * SEQ + kb;
        O[t] = wmma_f16(aP, load_frag(Vbase, hi), O[t]);
      }
    }
  }

  // ---- normalize and write head-concatenated output ----------------------
#pragma unroll
  for (int e = 0; e < 8; ++e) {
    float inv = 1.f / lrun[e];
    size_t obase = ((size_t)b * SEQ + (q0 + e + 8 * hi)) * DMODEL + h * HD;
#pragma unroll
    for (int t = 0; t < 4; ++t)
      attn[obase + t * 16 + lo] = (_Float16)(O[t][e] * inv);
  }
}

// ---------------------------------------------------------------------------
// Row-wise layernorm: torch semantics (unbiased std, eps added to std).
// ---------------------------------------------------------------------------
__global__ __launch_bounds__(256) void layernorm_kernel(
    const float* __restrict__ y, const float* __restrict__ g,
    const float* __restrict__ be, float* __restrict__ out) {
  const int row = blockIdx.x, tid = threadIdx.x;
  const float* x = y + (size_t)row * DMODEL;
  float s = 0.f, sq = 0.f;
  for (int i = tid; i < DMODEL; i += 256) {
    float v = x[i];
    s += v;
    sq += v * v;
  }
  __shared__ float rs[256], rq[256];
  rs[tid] = s;
  rq[tid] = sq;
  __syncthreads();
  for (int off = 128; off > 0; off >>= 1) {
    if (tid < off) {
      rs[tid] += rs[tid + off];
      rq[tid] += rq[tid + off];
    }
    __syncthreads();
  }
  float mean = rs[0] * (1.f / DMODEL);
  float var = fmaxf((rq[0] - (float)DMODEL * mean * mean) / (DMODEL - 1), 0.f);
  float denom = sqrtf(var) + 1e-3f;
  float* o = out + (size_t)row * DMODEL;
  for (int i = tid; i < DMODEL; i += 256)
    o[i] = (x[i] - mean) / denom * g[i] + be[i];
}

// ---------------------------------------------------------------------------
extern "C" void kernel_launch(void* const* d_in, const int* in_sizes, int n_in,
                              void* d_out, int out_size, void* d_ws,
                              size_t ws_size, hipStream_t stream) {
  const float* v = (const float*)d_in[0];
  const float* k = (const float*)d_in[1];
  const float* q = (const float*)d_in[2];
  const unsigned char* mask = (const unsigned char*)d_in[3];  // jax bool = 1B
  const float* w_q = (const float*)d_in[4];
  const float* w_k = (const float*)d_in[5];
  const float* w_v = (const float*)d_in[6];
  const float* w_o = (const float*)d_in[7];
  const float* b_o = (const float*)d_in[8];
  const float* gamma = (const float*)d_in[9];
  const float* beta = (const float*)d_in[10];
  float* out = (float*)d_out;

  // workspace layout (49 MB total)
  char* ws = (char*)d_ws;
  _Float16* qh = (_Float16*)(ws);                        //  8 MB [h][b][l][64]
  _Float16* kh = (_Float16*)(ws + ((size_t)8 << 20));    //  8 MB [h][b][l][64]
  _Float16* vt = (_Float16*)(ws + ((size_t)16 << 20));   //  8 MB [h][b][64][l]
  _Float16* attn = (_Float16*)(ws + ((size_t)24 << 20)); //  8 MB [b][l][1024]
  float* y = (float*)(ws + ((size_t)32 << 20));          // 16 MB pre-LN
  unsigned* mw = (unsigned*)(ws + ((size_t)48 << 20));   //  1 MB packed mask

  const int nWords = BATCH * SEQ * MWORDS;  // 262144
  mask_pack_kernel<<<nWords / 256, 256, 0, stream>>>(mask, mw);

  dim3 gproj(MTOT / 128, DMODEL / 64);
  proj_kernel<0><<<gproj, 256, 0, stream>>>(q, nullptr, w_q, qh, nullptr,
                                            nullptr, nullptr);
  proj_kernel<0><<<gproj, 256, 0, stream>>>(k, nullptr, w_k, kh, nullptr,
                                            nullptr, nullptr);
  proj_kernel<1><<<gproj, 256, 0, stream>>>(v, nullptr, w_v, vt, nullptr,
                                            nullptr, nullptr);
  flash_kernel<<<dim3(SEQ / 128, NHEAD, BATCH), 256, 0, stream>>>(qh, kh, vt,
                                                                  mw, attn);
  proj_kernel<2><<<gproj, 256, 0, stream>>>(nullptr, attn, w_o, nullptr, y,
                                            b_o, /*residual=*/k);
  layernorm_kernel<<<MTOT, 256, 0, stream>>>(y, gamma, beta, out);
  (void)in_sizes; (void)n_in; (void)out_size; (void)ws_size;
}
